// SSMBackbone_17514876633682
// MI455X (gfx1250) — matrix operand order repaired and checked by reading, not compiled, not run on hardware
//
#include <hip/hip_runtime.h>
#include <hip/hip_bf16.h>
#include <math.h>

#define B_SZ   4
#define T_LEN  512
#define H_DIM  256
#define N_DIM  64
#define L_NUM  4
#define M_ROWS (B_SZ * T_LEN)        // 2048
#define NCAT   (H_DIM + N_DIM)       // 320 : [Du | Bu] packed output width
#define CS     32                    // scan chunk size
#define NC     (T_LEN / CS)          // 16 chunks
#define LN_EPS 1e-5f

typedef __attribute__((ext_vector_type(16))) _Float16 v16h;
typedef __attribute__((ext_vector_type(8)))  _Float16 v8h;
typedef __attribute__((ext_vector_type(8)))  float    v8f;

#define X16_SZ    ((size_t)M_ROWS * H_DIM)
#define W_SZ      ((size_t)H_DIM * H_DIM)          // 65536
#define WCAT_SZ   ((size_t)NCAT * H_DIM)           // 81920 per layer
#define NN_SZ     ((size_t)N_DIM * N_DIM)          // 4096

__device__ __forceinline__ v16h ld16(const _Float16* p0, const _Float16* p1)
{
  v8h lo = *(const v8h*)p0;
  v8h hi = *(const v8h*)p1;
  return __builtin_shufflevector(lo, hi, 0,1,2,3,4,5,6,7,8,9,10,11,12,13,14,15);
}

// ---------------------------------------------------------------------------
// One-shot prep: convert x, Win, Wout to f16; pack [Dm_l ; Bm_l] -> Wcat_l f16
// ---------------------------------------------------------------------------
__global__ void prep_f16(const float* __restrict__ x,
                         const float* __restrict__ Win,
                         const float* __restrict__ Wout,
                         const float* __restrict__ Bm,
                         const float* __restrict__ Dm,
                         _Float16* __restrict__ x16,
                         _Float16* __restrict__ Win16,
                         _Float16* __restrict__ Wout16,
                         _Float16* __restrict__ Wcat16)
{
  size_t i = (size_t)blockIdx.x * blockDim.x + threadIdx.x;
  if (i < X16_SZ) { x16[i] = (_Float16)x[i]; return; }
  i -= X16_SZ;
  if (i < W_SZ)   { Win16[i] = (_Float16)Win[i]; return; }
  i -= W_SZ;
  if (i < W_SZ)   { Wout16[i] = (_Float16)Wout[i]; return; }
  i -= W_SZ;
  if (i < (size_t)L_NUM * WCAT_SZ) {
    const size_t l = i / WCAT_SZ;
    const size_t r = i % WCAT_SZ;
    float v;
    if (r < W_SZ) v = Dm[l * W_SZ + r];                           // rows 0..255  = D
    else          v = Bm[l * (size_t)N_DIM * H_DIM + (r - W_SZ)]; // rows 256..319 = B
    Wcat16[i] = (_Float16)v;
  }
}

// ---------------------------------------------------------------------------
// GEMM: Out[m,n] = sum_k A16[m,k]*W16[n,k] (+bias[n]); optional f16 mirror.
// Wave32 computes a 32(M) x 64(N) macro-tile: 2 A-fragments x 4 B-fragments,
// 8 accumulators, 8 independent WMMAs per k-step. Depth-1 software pipeline;
// __launch_bounds__(128,1) lets the backend keep both fragment buffers live.
// ---------------------------------------------------------------------------
__global__ __launch_bounds__(128, 1)
void gemm_wmma_f16(const _Float16* __restrict__ A16, // M x K(=H_DIM)
                   const _Float16* __restrict__ W16, // N x K
                   const float* __restrict__ bias,   // N or null
                   float* __restrict__ Out,          // M x N
                   _Float16* __restrict__ Out16,     // M x N or null
                   int N)
{
  const int K    = H_DIM;
  const int wave = threadIdx.x >> 5;
  const int lane = threadIdx.x & 31;
  const int r    = lane & 15;
  const int hi   = lane >> 4;

  const int tmBase = (blockIdx.x * 4 + wave) * 32;   // 32 rows per wave
  const int tn0    = blockIdx.y * 64;

  const _Float16* Ar0 = A16 + (size_t)(tmBase +  0 + r) * K;
  const _Float16* Ar1 = A16 + (size_t)(tmBase + 16 + r) * K;
  const _Float16* Wr0 = W16 + (size_t)(tn0 +  0 + r) * K;
  const _Float16* Wr1 = W16 + (size_t)(tn0 + 16 + r) * K;
  const _Float16* Wr2 = W16 + (size_t)(tn0 + 32 + r) * K;
  const _Float16* Wr3 = W16 + (size_t)(tn0 + 48 + r) * K;

  v8f acc00 = {}, acc01 = {}, acc02 = {}, acc03 = {};
  v8f acc10 = {}, acc11 = {}, acc12 = {}, acc13 = {};

  // prologue: fragments for kb = 0
  int  ao = hi * 8;
  int  ko = hi * 8;
  v16h a0 = ld16(Ar0 + ao, Ar0 + 16 + ao);
  v16h a1 = ld16(Ar1 + ao, Ar1 + 16 + ao);
  v16h b0 = ld16(Wr0 + ko, Wr0 + ko + 16);
  v16h b1 = ld16(Wr1 + ko, Wr1 + ko + 16);
  v16h b2 = ld16(Wr2 + ko, Wr2 + ko + 16);
  v16h b3 = ld16(Wr3 + ko, Wr3 + ko + 16);

#pragma unroll
  for (int kb = 0; kb < H_DIM; kb += 32) {
    v16h an0, an1, bn0, bn1, bn2, bn3;
    if (kb + 32 < H_DIM) {           // issue next-step loads before the WMMAs
      const int kn  = kb + 32;
      const int aon = kn + hi * 8;
      const int kon = kn + hi * 8;
      an0 = ld16(Ar0 + aon, Ar0 + 16 + aon);
      an1 = ld16(Ar1 + aon, Ar1 + 16 + aon);
      bn0 = ld16(Wr0 + kon, Wr0 + kon + 16);
      bn1 = ld16(Wr1 + kon, Wr1 + kon + 16);
      bn2 = ld16(Wr2 + kon, Wr2 + kon + 16);
      bn3 = ld16(Wr3 + kon, Wr3 + kon + 16);
    }
    acc00 = __builtin_amdgcn_wmma_f32_16x16x32_f16(false, a0, false, b0, (short)0, acc00, false, false);
    acc01 = __builtin_amdgcn_wmma_f32_16x16x32_f16(false, a0, false, b1, (short)0, acc01, false, false);
    acc02 = __builtin_amdgcn_wmma_f32_16x16x32_f16(false, a0, false, b2, (short)0, acc02, false, false);
    acc03 = __builtin_amdgcn_wmma_f32_16x16x32_f16(false, a0, false, b3, (short)0, acc03, false, false);
    acc10 = __builtin_amdgcn_wmma_f32_16x16x32_f16(false, a1, false, b0, (short)0, acc10, false, false);
    acc11 = __builtin_amdgcn_wmma_f32_16x16x32_f16(false, a1, false, b1, (short)0, acc11, false, false);
    acc12 = __builtin_amdgcn_wmma_f32_16x16x32_f16(false, a1, false, b2, (short)0, acc12, false, false);
    acc13 = __builtin_amdgcn_wmma_f32_16x16x32_f16(false, a1, false, b3, (short)0, acc13, false, false);
    if (kb + 32 < H_DIM) { a0 = an0; a1 = an1; b0 = bn0; b1 = bn1; b2 = bn2; b3 = bn3; }
  }

  v8f accs[2][4] = {{acc00, acc01, acc02, acc03}, {acc10, acc11, acc12, acc13}};
#pragma unroll
  for (int mt = 0; mt < 2; ++mt) {
    const int rowBase = tmBase + mt * 16;
#pragma unroll
    for (int nt = 0; nt < 4; ++nt) {
      const int   col = tn0 + nt * 16 + r;
      const float bv  = bias ? bias[col] : 0.0f;
#pragma unroll
      for (int i = 0; i < 8; ++i) {
        const float  v   = accs[mt][nt][i] + bv;
        const size_t off = (size_t)(rowBase + i + 8 * hi) * N + col;
        Out[off] = v;
        if (Out16) Out16[off] = (_Float16)v;
      }
    }
  }
}

// ---------------------------------------------------------------------------
// Cbar[l,n] = mean_h Cm[l,h,n]
// ---------------------------------------------------------------------------
__global__ void cbar_kernel(const float* __restrict__ Cm, float* __restrict__ Cbar)
{
  const int l = blockIdx.x;
  const int n = threadIdx.x;
  const float* C = Cm + (size_t)l * H_DIM * N_DIM;
  float s = 0.f;
  for (int h = 0; h < H_DIM; ++h) s += C[(size_t)h * N_DIM + n];
  Cbar[l * N_DIM + n] = s * (1.0f / H_DIM);
}

// ---------------------------------------------------------------------------
// A32[l] = Amat[l]^32 via 5 repeated squarings in LDS. 1 block / layer.
// ---------------------------------------------------------------------------
__global__ void apow32_kernel(const float* __restrict__ Amat, float* __restrict__ A32g)
{
  const int l = blockIdx.x;
  __shared__ float bufA[NN_SZ];
  __shared__ float bufB[NN_SZ];
  const float* A = Amat + (size_t)l * NN_SZ;
  for (int i = threadIdx.x; i < (int)NN_SZ; i += blockDim.x) bufA[i] = A[i];
  __syncthreads();

  float* cur = bufA;
  float* nxt = bufB;
  for (int it = 0; it < 5; ++it) {         // A^2, A^4, A^8, A^16, A^32
    const int r  = threadIdx.x >> 2;
    const int c0 = (threadIdx.x & 3) * 16;
    float out[16];
#pragma unroll
    for (int cc = 0; cc < 16; ++cc) out[cc] = 0.f;
    for (int m = 0; m < N_DIM; ++m) {
      const float av = cur[r * N_DIM + m];
#pragma unroll
      for (int cc = 0; cc < 16; ++cc) out[cc] += av * cur[m * N_DIM + c0 + cc];
    }
#pragma unroll
    for (int cc = 0; cc < 16; ++cc) nxt[r * N_DIM + c0 + cc] = out[cc];
    __syncthreads();
    float* t = cur; cur = nxt; nxt = t;
  }
  for (int i = threadIdx.x; i < (int)NN_SZ; i += blockDim.x)
    A32g[(size_t)l * NN_SZ + i] = cur[i];
}

// ---------------------------------------------------------------------------
// Phase B: per (batch, chunk) local reduction  f = sum_i A^(CS-1-i) Bu_i
// ---------------------------------------------------------------------------
__global__ void chunk_reduce(const float* __restrict__ Bu, int ld,
                             const float* __restrict__ Amat,
                             float* __restrict__ fbuf)     // (B, NC, N)
{
  const int c = blockIdx.x;
  const int b = blockIdx.y;
  const int n = threadIdx.x;

  __shared__ float sS[N_DIM];
  float Arow[N_DIM];
#pragma unroll 8
  for (int m = 0; m < N_DIM; ++m) Arow[m] = Amat[n * N_DIM + m];
  sS[n] = 0.f;
  __syncthreads();

  const float* BuC = Bu + ((size_t)b * T_LEN + (size_t)c * CS) * ld;
  float cur = BuC[n];
  float nv  = 0.f;
  for (int i = 0; i < CS; ++i) {
    const float nxt = (i + 1 < CS) ? BuC[(size_t)(i + 1) * ld + n] : 0.f;  // prefetch
    nv = cur;
#pragma unroll 8
    for (int m = 0; m < N_DIM; ++m) nv += Arow[m] * sS[m];
    __syncthreads();
    sS[n] = nv;
    cur   = nxt;
    __syncthreads();
  }
  fbuf[((size_t)b * NC + c) * N_DIM + n] = nv;
}

// ---------------------------------------------------------------------------
// Phase C: serial prefix over chunk boundaries: S_{c+1} = A^32 S_c + f_c.
// ---------------------------------------------------------------------------
__global__ void chunk_prefix(const float* __restrict__ A32,
                             const float* __restrict__ fbuf,
                             float* __restrict__ Sbuf)      // (B, NC, N)
{
  const int b = blockIdx.x;
  const int n = threadIdx.x;

  __shared__ float sS[N_DIM];
  float Prow[N_DIM];
#pragma unroll 8
  for (int m = 0; m < N_DIM; ++m) Prow[m] = A32[n * N_DIM + m];
  sS[n] = 0.f;
  __syncthreads();

  for (int c = 0; c < NC; ++c) {
    Sbuf[((size_t)b * NC + c) * N_DIM + n] = sS[n];
    float nv = fbuf[((size_t)b * NC + c) * N_DIM + n];
#pragma unroll 8
    for (int m = 0; m < N_DIM; ++m) nv += Prow[m] * sS[m];
    __syncthreads();
    sS[n] = nv;
    __syncthreads();
  }
}

// ---------------------------------------------------------------------------
// Phase D: interior replay from chunk-start state, emits c_t = <Cbar, s_t>.
// ---------------------------------------------------------------------------
__global__ void chunk_scan_out(const float* __restrict__ Bu, int ld,
                               const float* __restrict__ Amat,
                               const float* __restrict__ Cbar,
                               const float* __restrict__ Sbuf,
                               float* __restrict__ cOut)    // (B, T)
{
  const int c = blockIdx.x;
  const int b = blockIdx.y;
  const int n = threadIdx.x;

  __shared__ float sS[N_DIM];
  __shared__ float sC[N_DIM];
  float Arow[N_DIM];
#pragma unroll 8
  for (int m = 0; m < N_DIM; ++m) Arow[m] = Amat[n * N_DIM + m];
  const float cb = Cbar[n];

  sS[n] = Sbuf[((size_t)b * NC + c) * N_DIM + n];
  __syncthreads();

  const float* BuC = Bu + ((size_t)b * T_LEN + (size_t)c * CS) * ld;
  float*       cB  = cOut + (size_t)b * T_LEN + (size_t)c * CS;

  float cur = BuC[n];
  for (int i = 0; i < CS; ++i) {
    const float nxt = (i + 1 < CS) ? BuC[(size_t)(i + 1) * ld + n] : 0.f;  // prefetch
    float nv = cur;
#pragma unroll 8
    for (int m = 0; m < N_DIM; ++m) nv += Arow[m] * sS[m];
    sC[n] = cb * sS[n];
    __syncthreads();
    if (n == 0) {
      float s = 0.f;
#pragma unroll 8
      for (int m = 0; m < N_DIM; ++m) s += sC[m];
      cB[i] = s;
    }
    sS[n] = nv;
    cur   = nxt;
    __syncthreads();
  }
}

// ---------------------------------------------------------------------------
// wave32 block reduction (256 threads = 8 waves)
// ---------------------------------------------------------------------------
__device__ __forceinline__ float block_reduce_sum(float v, float* red)
{
#pragma unroll
  for (int m = 16; m >= 1; m >>= 1) v += __shfl_xor(v, m, 32);
  const int wid  = threadIdx.x >> 5;
  const int lane = threadIdx.x & 31;
  if (lane == 0) red[wid] = v;
  __syncthreads();
  if (wid == 0) {
    float t = (lane < 8) ? red[lane] : 0.f;
#pragma unroll
    for (int m = 4; m >= 1; m >>= 1) t += __shfl_xor(t, m, 32);
    if (lane == 0) red[0] = t;
  }
  __syncthreads();
  float tot = red[0];
  __syncthreads();
  return tot;
}

// ---------------------------------------------------------------------------
// y = GELU(Du[m,:] + c[m]) + h[m,:]; h = LN(y) in-place; also writes h16.
// ---------------------------------------------------------------------------
__global__ void fused_gelu_res_ln(const float* __restrict__ Du, int ld,
                                  const float* __restrict__ cIn,
                                  float* __restrict__ h,
                                  _Float16* __restrict__ h16)
{
  const int m = blockIdx.x;
  const int t = threadIdx.x;
  __shared__ float red[8];

  float x = Du[(size_t)m * ld + t] + cIn[m];
  float g = 0.5f * x * (1.0f + erff(x * 0.70710678118654752f));   // exact GELU
  float y = g + h[(size_t)m * H_DIM + t];

  float mean = block_reduce_sum(y, red) * (1.0f / H_DIM);
  float d    = y - mean;
  float var  = block_reduce_sum(d * d, red) * (1.0f / H_DIM);
  float o    = d * rsqrtf(var + LN_EPS);
  h  [(size_t)m * H_DIM + t] = o;
  h16[(size_t)m * H_DIM + t] = (_Float16)o;
}

// ---------------------------------------------------------------------------
// In-place row LayerNorm (+ f16 copy for the final GEMM)
// ---------------------------------------------------------------------------
__global__ void layernorm_rows(float* __restrict__ h, _Float16* __restrict__ h16)
{
  const int m = blockIdx.x;
  const int t = threadIdx.x;
  __shared__ float red[8];

  float y    = h[(size_t)m * H_DIM + t];
  float mean = block_reduce_sum(y, red) * (1.0f / H_DIM);
  float d    = y - mean;
  float var  = block_reduce_sum(d * d, red) * (1.0f / H_DIM);
  float o    = d * rsqrtf(var + LN_EPS);
  h  [(size_t)m * H_DIM + t] = o;
  h16[(size_t)m * H_DIM + t] = (_Float16)o;
}

// ---------------------------------------------------------------------------
extern "C" void kernel_launch(void* const* d_in, const int* in_sizes, int n_in,
                              void* d_out, int out_size, void* d_ws, size_t ws_size,
                              hipStream_t stream)
{
  (void)in_sizes; (void)n_in; (void)out_size; (void)ws_size;

  const float* x    = (const float*)d_in[0];
  const float* Win  = (const float*)d_in[1];
  const float* bin_ = (const float*)d_in[2];
  const float* Wout = (const float*)d_in[3];
  const float* bout = (const float*)d_in[4];
  const float* Amat = (const float*)d_in[5];
  const float* Bm   = (const float*)d_in[6];
  const float* Cm   = (const float*)d_in[7];
  const float* Dm   = (const float*)d_in[8];
  float* out = (float*)d_out;

  // ---- fp32 workspace ----
  float* ws   = (float*)d_ws;
  float* h    = ws;                                    // M*H
  float* DuBu = h    + (size_t)M_ROWS * H_DIM;         // M*NCAT
  float* cbuf = DuBu + (size_t)M_ROWS * NCAT;          // B*T
  float* Cb   = cbuf + (size_t)B_SZ * T_LEN;           // L*N
  float* A32g = Cb   + (size_t)L_NUM * N_DIM;          // L*N*N
  float* fbuf = A32g + (size_t)L_NUM * NN_SZ;          // B*NC*N
  float* Sbuf = fbuf + (size_t)B_SZ * NC * N_DIM;      // B*NC*N
  // ---- f16 workspace (fp32 region is a multiple of 4 floats -> 16B aligned) ----
  _Float16* f16ws  = (_Float16*)(Sbuf + (size_t)B_SZ * NC * N_DIM);
  _Float16* x16    = f16ws;
  _Float16* h16    = x16    + X16_SZ;
  _Float16* Win16  = h16    + X16_SZ;
  _Float16* Wout16 = Win16  + W_SZ;
  _Float16* Wcat16 = Wout16 + W_SZ;                    // L * NCAT * H

  {
    const size_t total  = X16_SZ + 2 * W_SZ + (size_t)L_NUM * WCAT_SZ;
    const int    blocks = (int)((total + 255) / 256);
    prep_f16<<<dim3(blocks), dim3(256), 0, stream>>>(x, Win, Wout, Bm, Dm,
                                                     x16, Win16, Wout16, Wcat16);
  }
  cbar_kernel<<<dim3(L_NUM), dim3(N_DIM), 0, stream>>>(Cm, Cb);
  apow32_kernel<<<dim3(L_NUM), dim3(256), 0, stream>>>(Amat, A32g);

  // h = x @ Win^T + bin   (each block covers 128 rows x 64 cols)
  gemm_wmma_f16<<<dim3(M_ROWS / 128, H_DIM / 64), dim3(128), 0, stream>>>(
      x16, Win16, bin_, h, h16, H_DIM);

  for (int l = 0; l < L_NUM; ++l) {
    const float* Al = Amat + (size_t)l * NN_SZ;
    // [Du | Bu] = h @ [D^T | B^T]   (M x 320)
    gemm_wmma_f16<<<dim3(M_ROWS / 128, NCAT / 64), dim3(128), 0, stream>>>(
        h16, Wcat16 + (size_t)l * WCAT_SZ, nullptr, DuBu, nullptr, NCAT);
    // blocked parallel scan over the Bu slice (cols 256..319, stride NCAT)
    chunk_reduce<<<dim3(NC, B_SZ), dim3(N_DIM), 0, stream>>>(
        DuBu + H_DIM, NCAT, Al, fbuf);
    chunk_prefix<<<dim3(B_SZ), dim3(N_DIM), 0, stream>>>(
        A32g + (size_t)l * NN_SZ, fbuf, Sbuf);
    chunk_scan_out<<<dim3(NC, B_SZ), dim3(N_DIM), 0, stream>>>(
        DuBu + H_DIM, NCAT, Al, Cb + l * N_DIM, Sbuf, cbuf);
    // h = LN(GELU(Du + c) + h)
    fused_gelu_res_ln<<<dim3(M_ROWS), dim3(H_DIM), 0, stream>>>(
        DuBu, NCAT, cbuf, h, h16);
  }

  layernorm_rows<<<dim3(M_ROWS), dim3(H_DIM), 0, stream>>>(h, h16);
  gemm_wmma_f16<<<dim3(M_ROWS / 128, H_DIM / 64), dim3(128), 0, stream>>>(
      h16, Wout16, bout, out, nullptr, H_DIM);
}
